// VITBEncoder_19284403159384
// MI455X (gfx1250) — matrix operand order repaired and checked by reading, not compiled, not run on hardware
//
#include <hip/hip_runtime.h>
#include <math.h>

typedef _Float16 half16 __attribute__((ext_vector_type(16)));
typedef _Float16 half8  __attribute__((ext_vector_type(8)));
typedef float    float8 __attribute__((ext_vector_type(8)));

#define EDIM 768
#define NHEAD 12
#define HDIM 64
#define TOK 2048           // B(2) * 32 * 32 tokens
#define GRID 32

// ---------------- WMMA f16 fragment loaders (per CDNA5 ISA layouts) ----------------
// Unpredicated: row/col clamped (garbage only reaches rows/cols the epilogue
// never stores); K is a multiple of 32 by construction.

// A 16x32 f16, row-major f16 source: lane holds row m=base+(L&15);
// K = (j&7) + 8*(L>>4) + 16*(j>>3)  -> two contiguous 8-half (16B) chunks.
__device__ __forceinline__ half16 frag_a16(const _Float16* __restrict__ p, int rowBase,
                                           int lda, int kBase, int rowMax) {
  const int lane = threadIdx.x & 31;
  int r = rowBase + (lane & 15);
  if (r >= rowMax) r = rowMax - 1;
  const _Float16* base = p + (long)r * lda + kBase + ((lane >> 4) << 3);
  half8 c0 = *(const half8*)(base);
  half8 c1 = *(const half8*)(base + 16);
  half16 f;
#pragma unroll
  for (int j = 0; j < 8; ++j) { f[j] = c0[j]; f[j + 8] = c1[j]; }
  return f;
}

// B 32x16 f16, N-major f16 source: lane holds col n=base+(L&15);
// K = (j&7) + 8*(L>>4) + 16*(j>>3)  -> two contiguous 8-half (16B) chunks.
__device__ __forceinline__ half16 frag_b16(const _Float16* __restrict__ p, int nBase,
                                           int ldb, int kBase, int nMax) {
  const int lane = threadIdx.x & 31;
  int n = nBase + (lane & 15);
  if (n >= nMax) n = nMax - 1;
  const _Float16* base = p + (long)n * ldb + kBase + ((lane >> 4) << 3);
  half8 c0 = *(const half8*)(base);
  half8 c1 = *(const half8*)(base + 16);
  half16 f;
#pragma unroll
  for (int j = 0; j < 8; ++j) { f[j] = c0[j]; f[j + 8] = c1[j]; }
  return f;
}

// ---------------- Generic batched WMMA GEMM ----------------
// C[z] = act(alpha * A[z] @ B[z]^T(N-major) + bias) (+ res[z])
// A: f16 row-major, B: f16 N-major, C: f32 or f16 (OUT16).
// Block: 256 thr = 8 waves; wave tile 32x64 (2 A frags x 4 B frags, 8 WMMA/K-step),
// block tile 128x128 (4 waves M x 2 waves N). K%32==0, K>=64.
// Fragments are software-pipelined: next K-step's loads issue before the
// current step's WMMAs so the matrix pipe overlaps VMEM.
template <int ACT, int RES, int OUT16>
__global__ __launch_bounds__(256) void gemm_wmma_k(
    const _Float16* __restrict__ A, const _Float16* __restrict__ B,
    const float* __restrict__ bias, const float* __restrict__ res,
    void* __restrict__ Cv, int M, int N, int K,
    int lda, int ldb, int ldc, long sA, long sB, long sC, float alpha) {
  const int z = blockIdx.z;
  A += (long)z * sA;
  B += (long)z * sB;
  const float* Rp = RES ? (res + (long)z * sC) : nullptr;

  const int wave = threadIdx.x >> 5;
  const int lane = threadIdx.x & 31;
  const int wm = wave & 3, wn = wave >> 2;
  const int mBase = blockIdx.y * 128 + wm * 32;
  const int nBase = blockIdx.x * 128 + wn * 64;

  float8 acc[2][4] = {};

  half16 a0 = frag_a16(A, mBase + 0,  lda, 0, M);
  half16 a1 = frag_a16(A, mBase + 16, lda, 0, M);
  half16 b0 = frag_b16(B, nBase + 0,  ldb, 0, N);
  half16 b1 = frag_b16(B, nBase + 16, ldb, 0, N);
  half16 b2 = frag_b16(B, nBase + 32, ldb, 0, N);
  half16 b3 = frag_b16(B, nBase + 48, ldb, 0, N);

  for (int kB = 32; kB < K; kB += 32) {
    half16 na0 = frag_a16(A, mBase + 0,  lda, kB, M);
    half16 na1 = frag_a16(A, mBase + 16, lda, kB, M);
    half16 nb0 = frag_b16(B, nBase + 0,  ldb, kB, N);
    half16 nb1 = frag_b16(B, nBase + 16, ldb, kB, N);
    half16 nb2 = frag_b16(B, nBase + 32, ldb, kB, N);
    half16 nb3 = frag_b16(B, nBase + 48, ldb, kB, N);

    acc[0][0] = __builtin_amdgcn_wmma_f32_16x16x32_f16(false, a0, false, b0, (short)0, acc[0][0], false, false);
    acc[0][1] = __builtin_amdgcn_wmma_f32_16x16x32_f16(false, a0, false, b1, (short)0, acc[0][1], false, false);
    acc[0][2] = __builtin_amdgcn_wmma_f32_16x16x32_f16(false, a0, false, b2, (short)0, acc[0][2], false, false);
    acc[0][3] = __builtin_amdgcn_wmma_f32_16x16x32_f16(false, a0, false, b3, (short)0, acc[0][3], false, false);
    acc[1][0] = __builtin_amdgcn_wmma_f32_16x16x32_f16(false, a1, false, b0, (short)0, acc[1][0], false, false);
    acc[1][1] = __builtin_amdgcn_wmma_f32_16x16x32_f16(false, a1, false, b1, (short)0, acc[1][1], false, false);
    acc[1][2] = __builtin_amdgcn_wmma_f32_16x16x32_f16(false, a1, false, b2, (short)0, acc[1][2], false, false);
    acc[1][3] = __builtin_amdgcn_wmma_f32_16x16x32_f16(false, a1, false, b3, (short)0, acc[1][3], false, false);

    a0 = na0; a1 = na1; b0 = nb0; b1 = nb1; b2 = nb2; b3 = nb3;
  }

  acc[0][0] = __builtin_amdgcn_wmma_f32_16x16x32_f16(false, a0, false, b0, (short)0, acc[0][0], false, false);
  acc[0][1] = __builtin_amdgcn_wmma_f32_16x16x32_f16(false, a0, false, b1, (short)0, acc[0][1], false, false);
  acc[0][2] = __builtin_amdgcn_wmma_f32_16x16x32_f16(false, a0, false, b2, (short)0, acc[0][2], false, false);
  acc[0][3] = __builtin_amdgcn_wmma_f32_16x16x32_f16(false, a0, false, b3, (short)0, acc[0][3], false, false);
  acc[1][0] = __builtin_amdgcn_wmma_f32_16x16x32_f16(false, a1, false, b0, (short)0, acc[1][0], false, false);
  acc[1][1] = __builtin_amdgcn_wmma_f32_16x16x32_f16(false, a1, false, b1, (short)0, acc[1][1], false, false);
  acc[1][2] = __builtin_amdgcn_wmma_f32_16x16x32_f16(false, a1, false, b2, (short)0, acc[1][2], false, false);
  acc[1][3] = __builtin_amdgcn_wmma_f32_16x16x32_f16(false, a1, false, b3, (short)0, acc[1][3], false, false);

  const int nl = lane & 15, hi = lane >> 4;
#pragma unroll
  for (int s = 0; s < 2; ++s) {
#pragma unroll
    for (int t = 0; t < 4; ++t) {
      int n = nBase + 16 * t + nl;
      if (n >= N) continue;
      float bv = bias ? bias[n] : 0.0f;
#pragma unroll
      for (int r = 0; r < 8; ++r) {
        int m = mBase + s * 16 + r + 8 * hi;
        if (m >= M) continue;
        float c = acc[s][t][r] * alpha + bv;
        if (ACT == 1) c = 0.5f * c * (1.0f + erff(c * 0.70710678118654752f));
        if (RES) c += Rp[(long)m * ldc + n];
        if (OUT16) ((_Float16*)Cv)[(long)z * sC + (long)m * ldc + n] = (_Float16)c;
        else       ((float*)Cv)[(long)z * sC + (long)m * ldc + n] = c;
      }
    }
  }
}

// ---------------- Operand packing ----------------
// K-major f32 weight (KxN) -> N-major f16 (NxK)
__global__ void pack_wT_k(const float* __restrict__ B, _Float16* __restrict__ out,
                          int K, int N) {
  long idx = (long)blockIdx.x * blockDim.x + threadIdx.x;
  if (idx >= (long)K * N) return;
  int n = (int)(idx / K), k = (int)(idx % K);
  out[idx] = (_Float16)B[(long)k * N + n];
}

// straight f32 -> f16 convert (for already N-major weights / activations)
__global__ void cvt16_k(const float* __restrict__ in, _Float16* __restrict__ out, long total) {
  long idx = (long)blockIdx.x * blockDim.x + threadIdx.x;
  if (idx >= total) return;
  out[idx] = (_Float16)in[idx];
}

// ---------------- Elementwise / reduction kernels ----------------
__global__ void im2col_k(const float* __restrict__ x, _Float16* __restrict__ out) {
  long idx = (long)blockIdx.x * blockDim.x + threadIdx.x;
  if (idx >= (long)TOK * EDIM) return;
  int t = (int)(idx / EDIM), k2 = (int)(idx % EDIM);
  int b = t >> 10, r = t & 1023, gy = r >> 5, gx = r & 31;
  int c = k2 >> 8, pr = k2 & 255, py = pr >> 4, px = pr & 15;
  out[idx] = (_Float16)x[((long)(b * 3 + c) * 512 + gy * 16 + py) * 512 + gx * 16 + px];
}

__global__ void addpos_k(float* __restrict__ x, const float* __restrict__ pos) {
  long idx = (long)blockIdx.x * blockDim.x + threadIdx.x;
  if (idx >= (long)TOK * EDIM) return;
  x[idx] += pos[idx % ((long)1024 * EDIM)];
}

// f32 in -> f16 normalized out
__global__ void layernorm_k(const float* __restrict__ x, const float* __restrict__ w,
                            const float* __restrict__ b, _Float16* __restrict__ out) {
  const int t = blockIdx.x;
  const float* row = x + (long)t * EDIM;
  __shared__ float red[256];
  int tid = threadIdx.x;
  float s = 0.0f;
  for (int i = tid; i < EDIM; i += 256) s += row[i];
  red[tid] = s; __syncthreads();
  for (int o = 128; o > 0; o >>= 1) { if (tid < o) red[tid] += red[tid + o]; __syncthreads(); }
  float mu = red[0] / (float)EDIM;
  __syncthreads();
  float v = 0.0f;
  for (int i = tid; i < EDIM; i += 256) { float d = row[i] - mu; v += d * d; }
  red[tid] = v; __syncthreads();
  for (int o = 128; o > 0; o >>= 1) { if (tid < o) red[tid] += red[tid + o]; __syncthreads(); }
  float rstd = rsqrtf(red[0] / (float)EDIM + 1e-6f);
  _Float16* orow = out + (long)t * EDIM;
  for (int i = tid; i < EDIM; i += 256)
    orow[i] = (_Float16)((row[i] - mu) * rstd * w[i] + b[i]);
}

// qkv16 (TOK x 2304) -> staged q,k [nseq,Pp,64] and v transposed [nseq,64,Pp]
__global__ void qkv_scatter_k(const _Float16* __restrict__ qkv, const float* __restrict__ qkvb,
                              _Float16* __restrict__ q, _Float16* __restrict__ k,
                              _Float16* __restrict__ vT, int windowed, int nseq, int Pp, int valid) {
  long idx = (long)blockIdx.x * blockDim.x + threadIdx.x;
  long total = (long)nseq * Pp * HDIM;
  if (idx >= total) return;
  int seq = (int)(idx / ((long)Pp * HDIM));
  int rem = (int)(idx % ((long)Pp * HDIM));
  int p = rem / HDIM, d = rem % HDIM;
  int hh = seq % NHEAD;
  _Float16 qv = (_Float16)0.0f, kv = (_Float16)0.0f, vv = (_Float16)0.0f;
  if (windowed) {
    int w = seq / NHEAD, b = w / 9, wr = (w % 9) / 3, wc = w % 3;
    if (p < valid) {
      int wy = p / 14, wx = p % 14;
      int gy = wr * 14 + wy, gx = wc * 14 + wx;
      if (gy < GRID && gx < GRID) {
        const _Float16* rw = qkv + ((long)b * 1024 + gy * GRID + gx) * 2304;
        qv = rw[hh * 64 + d]; kv = rw[768 + hh * 64 + d]; vv = rw[1536 + hh * 64 + d];
      } else {  // zero-padded token -> qkv = bias only
        qv = (_Float16)qkvb[hh * 64 + d];
        kv = (_Float16)qkvb[768 + hh * 64 + d];
        vv = (_Float16)qkvb[1536 + hh * 64 + d];
      }
    }
  } else {
    int b = seq / NHEAD;
    const _Float16* rw = qkv + ((long)b * 1024 + p) * 2304;
    qv = rw[hh * 64 + d]; kv = rw[768 + hh * 64 + d]; vv = rw[1536 + hh * 64 + d];
  }
  q[idx] = qv;
  k[idx] = kv;
  vT[((long)seq * HDIM + d) * Pp + p] = vv;   // N-major for probs@v GEMM
}

// rh[seq][p][j] = q[seq][p][:] . rel_h[qy-j+W-1][:]   (and rw with qx)
__global__ void relpos_k(const _Float16* __restrict__ q, const float* __restrict__ relh,
                         const float* __restrict__ relw, float* __restrict__ rh,
                         float* __restrict__ rw, int nseq, int Pp, int valid, int Wd) {
  long idx = (long)blockIdx.x * blockDim.x + threadIdx.x;
  long total = (long)nseq * Pp * Wd;
  if (idx >= total) return;
  int seq = (int)(idx / ((long)Pp * Wd));
  int rem = (int)(idx % ((long)Pp * Wd));
  int p = rem / Wd, j = rem % Wd;
  float rhv = 0.0f, rwv = 0.0f;
  if (p < valid) {
    int qy = p / Wd, qx = p % Wd;
    const _Float16* qrow = q + ((long)seq * Pp + p) * HDIM;
    const float* hrow = relh + (long)(qy - j + Wd - 1) * HDIM;
    const float* wrow = relw + (long)(qx - j + Wd - 1) * HDIM;
    for (int d = 0; d < HDIM; ++d) {
      float qd = (float)qrow[d];
      rhv += qd * hrow[d];
      rwv += qd * wrow[d];
    }
  }
  rh[idx] = rhv; rw[idx] = rwv;
}

// S[seq][i][j] += rh[seq][i][j/Wd] + rw[seq][i][j%Wd]   (j < valid)
__global__ void relbias_k(float* __restrict__ S, const float* __restrict__ rh,
                          const float* __restrict__ rw, int nseq, int Pp, int valid, int Wd) {
  long idx = (long)blockIdx.x * blockDim.x + threadIdx.x;
  long total = (long)nseq * Pp * valid;
  if (idx >= total) return;
  int seq = (int)(idx / ((long)Pp * valid));
  int rem = (int)(idx % ((long)Pp * valid));
  int i = rem / valid, j = rem % valid;
  int ky = j / Wd, kx = j % Wd;
  long base = (long)seq * Pp + i;
  S[base * Pp + j] += rh[base * Wd + ky] + rw[base * Wd + kx];
}

// f32 scores row -> f16 prob row (pad cols zeroed)
__global__ void softmax_k(const float* __restrict__ S, _Float16* __restrict__ P,
                          int ld, int valid) {
  const float* row = S + (long)blockIdx.x * ld;
  _Float16* prow = P + (long)blockIdx.x * ld;
  __shared__ float red[256];
  int tid = threadIdx.x;
  float mx = -3.0e38f;
  for (int i = tid; i < valid; i += 256) mx = fmaxf(mx, row[i]);
  red[tid] = mx; __syncthreads();
  for (int o = 128; o > 0; o >>= 1) { if (tid < o) red[tid] = fmaxf(red[tid], red[tid + o]); __syncthreads(); }
  float m = red[0];
  __syncthreads();
  float s = 0.0f;
  for (int i = tid; i < valid; i += 256) s += __expf(row[i] - m);
  red[tid] = s; __syncthreads();
  for (int o = 128; o > 0; o >>= 1) { if (tid < o) red[tid] += red[tid + o]; __syncthreads(); }
  float inv = 1.0f / red[0];
  for (int i = tid; i < valid; i += 256) prow[i] = (_Float16)(__expf(row[i] - m) * inv);
  for (int i = valid + tid; i < ld; i += 256) prow[i] = (_Float16)0.0f;  // mask pad cols
}

// staged attn out [nseq, Pp, 64] f16 -> token layout h16[TOK, 768] (crops window pad)
__global__ void attn_gather_k(const _Float16* __restrict__ ao, _Float16* __restrict__ h,
                              int windowed, int Pp) {
  long idx = (long)blockIdx.x * blockDim.x + threadIdx.x;
  if (idx >= (long)TOK * EDIM) return;
  int t = (int)(idx / EDIM), e = (int)(idx % EDIM);
  int hh = e >> 6, d = e & 63;
  int b = t >> 10, r = t & 1023, gy = r >> 5, gx = r & 31;
  long seq, p;
  if (windowed) {
    int wr = gy / 14, wc = gx / 14, wy = gy % 14, wx = gx % 14;
    seq = (long)(b * 9 + wr * 3 + wc) * NHEAD + hh;
    p = wy * 14 + wx;
  } else {
    seq = (long)b * NHEAD + hh;
    p = r;
  }
  h[idx] = ao[(seq * Pp + p) * HDIM + d];
}

// (TOK x 256) f32 -> (B, 256, 32, 32) + out_b
__global__ void head_out_k(const float* __restrict__ tmp, const float* __restrict__ ob,
                           float* __restrict__ out) {
  long idx = (long)blockIdx.x * blockDim.x + threadIdx.x;
  long total = (long)2 * 256 * 1024;
  if (idx >= total) return;
  int b = (int)(idx / (256 * 1024));
  int rem = (int)(idx % (256 * 1024));
  int o = rem / 1024, pr = rem % 1024;
  out[idx] = tmp[((long)b * 1024 + pr) * 256 + o] + ob[o];
}

// ---------------- host-side helpers ----------------
static inline void launch_gemm(hipStream_t st, const _Float16* A, const _Float16* B,
                               const float* bias, const float* res, void* C,
                               int M, int N, int K, int lda, int ldb, int ldc,
                               long sA, long sB, long sC, int batch, float alpha,
                               int act, int resf, int out16) {
  dim3 g((N + 127) / 128, (M + 127) / 128, batch), blk(256);
#define GO(ACT, RES, O16) \
  gemm_wmma_k<ACT, RES, O16><<<g, blk, 0, st>>>(A, B, bias, res, C, M, N, K, lda, ldb, ldc, sA, sB, sC, alpha)
  if (act)          GO(1, 0, 1);   // fc1: GELU, f16 out
  else if (resf)    GO(0, 1, 0);   // proj/fc2: residual, f32 out
  else if (out16)   GO(0, 0, 1);   // qkv / probs@v: f16 out
  else              GO(0, 0, 0);   // patch / scores / head: f32 out
#undef GO
}

static inline dim3 eltgrid(long total) { return dim3((unsigned)((total + 255) / 256)); }

extern "C" void kernel_launch(void* const* d_in, const int* in_sizes, int n_in,
                              void* d_out, int out_size, void* d_ws, size_t ws_size,
                              hipStream_t stream) {
  const float* X    = (const float*)d_in[0];
  const float* PW   = (const float*)d_in[1];
  const float* PB   = (const float*)d_in[2];
  const float* POS  = (const float*)d_in[3];
  const float* LN1W = (const float*)d_in[4];
  const float* LN1B = (const float*)d_in[5];
  const float* QKVW = (const float*)d_in[6];
  const float* QKVB = (const float*)d_in[7];
  const float* PRJW = (const float*)d_in[8];
  const float* PRJB = (const float*)d_in[9];
  const float* RHW  = (const float*)d_in[10];
  const float* RWW  = (const float*)d_in[11];
  const float* RHG  = (const float*)d_in[12];
  const float* RWG  = (const float*)d_in[13];
  const float* LN2W = (const float*)d_in[14];
  const float* LN2B = (const float*)d_in[15];
  const float* FC1W = (const float*)d_in[16];
  const float* FC1B = (const float*)d_in[17];
  const float* FC2W = (const float*)d_in[18];
  const float* FC2B = (const float*)d_in[19];
  const float* OUTW = (const float*)d_in[20];
  const float* OUTB = (const float*)d_in[21];
  (void)in_sizes; (void)n_in; (void)out_size; (void)ws_size;

  char* wsb = (char*)d_ws;
  const long QSZ = 216L * 224 * 64;      // staged q/k/vT/ao elems (window case, Pp=224)
  const long SSZ = 24L * 1024 * 1024;    // scores elems (global 25.2M > window 10.8M)
  const long RSZ = 24L * 1024 * 32;      // rel bias elems
  size_t off = 0;
  auto take = [&](long elems, size_t esz) {
    size_t o = off;
    off += ((size_t)elems * esz + 255) & ~(size_t)255;
    return o;
  };
  size_t b_x   = take((long)TOK * EDIM, 4);   // f32 residual stream
  size_t b_h16 = take((long)TOK * EDIM, 2);   // f16 activations (LN out / attn out / head in)
  size_t b_qkv = take((long)TOK * 2304, 2);   // f16 qkv
  size_t b_q   = take(QSZ, 2);
  size_t b_k   = take(QSZ, 2);
  size_t b_vT  = take(QSZ, 2);
  size_t b_ao  = take(QSZ, 2);
  size_t b_s   = take(SSZ, 4);                // f32 scores; reused as f32 head tmp
  size_t b_p   = take(SSZ, 2);                // f16 probs
  size_t b_rh  = take(RSZ, 4);
  size_t b_rw  = take(RSZ, 4);
  size_t b_mid = take((long)TOK * 3072, 2);   // f16 MLP mid; reused as f16 im2col
  size_t b_pk  = take(3072L * 768, 2);        // f16 packed weight (max fc shape)

  float*     xb   = (float*)(wsb + b_x);
  _Float16*  h16  = (_Float16*)(wsb + b_h16);
  _Float16*  qkv16= (_Float16*)(wsb + b_qkv);
  _Float16*  q16  = (_Float16*)(wsb + b_q);
  _Float16*  k16  = (_Float16*)(wsb + b_k);
  _Float16*  vT16 = (_Float16*)(wsb + b_vT);
  _Float16*  ao16 = (_Float16*)(wsb + b_ao);
  float*     sco  = (float*)(wsb + b_s);
  _Float16*  prb  = (_Float16*)(wsb + b_p);
  float*     rhb  = (float*)(wsb + b_rh);
  float*     rwb  = (float*)(wsb + b_rw);
  _Float16*  mid16= (_Float16*)(wsb + b_mid);
  _Float16*  pk   = (_Float16*)(wsb + b_pk);
  _Float16*  im16 = mid16;                    // reuse
  float*     hdt  = sco;                      // reuse

  // ---- patch embed: im2col(f16) + GEMM + bias, then pos embed ----
  im2col_k<<<eltgrid((long)TOK * EDIM), 256, 0, stream>>>(X, im16);
  cvt16_k<<<eltgrid((long)EDIM * EDIM), 256, 0, stream>>>(PW, pk, (long)EDIM * EDIM);  // already N-major
  launch_gemm(stream, im16, pk, PB, nullptr, xb,
              TOK, EDIM, EDIM, EDIM, EDIM, EDIM, 0, 0, 0, 1, 1.0f, 0, 0, 0);
  addpos_k<<<eltgrid((long)TOK * EDIM), 256, 0, stream>>>(xb, POS);

  int win_i = 0, glob_i = 0;
  for (int i = 0; i < 12; ++i) {
    const bool windowed = !(i == 2 || i == 5 || i == 8 || i == 11);
    const int nseq  = windowed ? 216 : 24;
    const int Pp    = windowed ? 224 : 1024;   // padded seq len (multiple of 32)
    const int valid = windowed ? 196 : 1024;
    const int Wd    = windowed ? 14 : 32;
    const float* relh = windowed ? RHW + (long)win_i * 27 * HDIM : RHG + (long)glob_i * 63 * HDIM;
    const float* relw = windowed ? RWW + (long)win_i * 27 * HDIM : RWG + (long)glob_i * 63 * HDIM;
    const float* qkvb = QKVB + (long)i * 2304;

    // LN1 -> f16
    layernorm_k<<<TOK, 256, 0, stream>>>(xb, LN1W + i * EDIM, LN1B + i * EDIM, h16);
    // qkv projection (pack weight, f16 out)
    pack_wT_k<<<eltgrid((long)EDIM * 2304), 256, 0, stream>>>(QKVW + (long)i * EDIM * 2304, pk, EDIM, 2304);
    launch_gemm(stream, h16, pk, qkvb, nullptr, qkv16,
                TOK, 2304, EDIM, EDIM, EDIM, 2304, 0, 0, 0, 1, 1.0f, 0, 0, 1);
    // scatter into per-(window/head) sequences (v stored transposed)
    qkv_scatter_k<<<eltgrid((long)nseq * Pp * HDIM), 256, 0, stream>>>(
        qkv16, qkvb, q16, k16, vT16, windowed ? 1 : 0, nseq, Pp, valid);
    // decomposed rel-pos dot products
    relpos_k<<<eltgrid((long)nseq * Pp * Wd), 256, 0, stream>>>(
        q16, relh, relw, rhb, rwb, nseq, Pp, valid, Wd);
    // scores = scale * q @ k^T  (k16 is N-major by construction)
    launch_gemm(stream, q16, k16, nullptr, nullptr, sco,
                Pp, Pp, HDIM, HDIM, HDIM, Pp,
                (long)Pp * HDIM, (long)Pp * HDIM, (long)Pp * Pp, nseq, 0.125f, 0, 0, 0);
    // + rel bias
    relbias_k<<<eltgrid((long)nseq * Pp * valid), 256, 0, stream>>>(
        sco, rhb, rwb, nseq, Pp, valid, Wd);
    // softmax -> f16 probs (masks pad columns)
    softmax_k<<<nseq * Pp, 256, 0, stream>>>(sco, prb, Pp, valid);
    // out = probs @ v  (vT16 is N-major; K = Pp)
    launch_gemm(stream, prb, vT16, nullptr, nullptr, ao16,
                Pp, HDIM, Pp, Pp, Pp, HDIM,
                (long)Pp * Pp, (long)Pp * HDIM, (long)Pp * HDIM, nseq, 1.0f, 0, 0, 1);
    // gather heads back to token layout (crop window padding)
    attn_gather_k<<<eltgrid((long)TOK * EDIM), 256, 0, stream>>>(ao16, h16,
                                                                 windowed ? 1 : 0, Pp);
    // proj + residual (f32 out into residual stream)
    pack_wT_k<<<eltgrid((long)EDIM * EDIM), 256, 0, stream>>>(PRJW + (long)i * EDIM * EDIM, pk, EDIM, EDIM);
    launch_gemm(stream, h16, pk, PRJB + i * EDIM, xb, xb,
                TOK, EDIM, EDIM, EDIM, EDIM, EDIM, 0, 0, 0, 1, 1.0f, 0, 1, 0);
    // LN2 -> f16
    layernorm_k<<<TOK, 256, 0, stream>>>(xb, LN2W + i * EDIM, LN2B + i * EDIM, h16);
    // fc1 + GELU (f16 out)
    pack_wT_k<<<eltgrid((long)EDIM * 3072), 256, 0, stream>>>(FC1W + (long)i * EDIM * 3072, pk, EDIM, 3072);
    launch_gemm(stream, h16, pk, FC1B + (long)i * 3072, nullptr, mid16,
                TOK, 3072, EDIM, EDIM, EDIM, 3072, 0, 0, 0, 1, 1.0f, 1, 0, 1);
    // fc2 + residual
    pack_wT_k<<<eltgrid((long)3072 * EDIM), 256, 0, stream>>>(FC2W + (long)i * 3072 * EDIM, pk, 3072, EDIM);
    launch_gemm(stream, mid16, pk, FC2B + i * EDIM, xb, xb,
                TOK, EDIM, 3072, 3072, 3072, EDIM, 0, 0, 0, 1, 1.0f, 0, 1, 0);

    if (windowed) ++win_i; else ++glob_i;
  }

  // ---- output head: x @ out_w^T (out_w already N-major), then NCHW + bias ----
  cvt16_k<<<eltgrid((long)TOK * EDIM), 256, 0, stream>>>(xb, h16, (long)TOK * EDIM);
  cvt16_k<<<eltgrid(256L * EDIM), 256, 0, stream>>>(OUTW, pk, 256L * EDIM);
  launch_gemm(stream, h16, pk, nullptr, nullptr, hdt,
              TOK, 256, EDIM, EDIM, EDIM, 256, 0, 0, 0, 1, 1.0f, 0, 0, 0);
  head_out_k<<<eltgrid((long)2 * 256 * 1024), 256, 0, stream>>>(hdt, OUTB, (float*)d_out);
}
